// Encoder_4578435138214
// MI455X (gfx1250) — matrix operand order repaired and checked by reading, not compiled, not run on hardware
//
#include <hip/hip_runtime.h>
#include <hip/hip_bf16.h>

// ---------------------------------------------------------------------------
// Sparse-masked 3D CNN encoder for MI455X (gfx1250).
// Heavy convs (Cin>=64) run as implicit GEMM on V_WMMA_F32_16X16X32_F16
// (f16 A/B, f32 accum). Light convs (Cin in {1,16}) use scalar VALU FMA.
// All conv shapes compile-time (pow2 dims -> shift/mask decode, fully
// unrolled K-chunk loop). Each wave owns NTPW adjacent 16x16 N-tiles and
// reuses one A fragment across them (higher FLOP/byte, independent WMMAs).
// ---------------------------------------------------------------------------

typedef __attribute__((ext_vector_type(16))) _Float16 v16h;
typedef __attribute__((ext_vector_type(8)))  float    v8f;

union F16Frag { v16h v; uint4 q[2]; };

// ---------------- problem constants ----------------
static constexpr int  BN_ = 4;
static constexpr int  D0 = 128, D1 = 64, D2 = 32, D3 = 16;
static constexpr long S0 = (long)BN_ * D0 * D0 * D0;  // 8388608
static constexpr long S1 = (long)BN_ * D1 * D1 * D1;  // 1048576
static constexpr long S2 = (long)BN_ * D2 * D2 * D2;  // 131072
static constexpr long S3 = (long)BN_ * D3 * D3 * D3;  // 16384

// ---------------- workspace layout (256B aligned) ----------------
static constexpr size_t alup(size_t x) { return (x + 255) & ~(size_t)255; }
static constexpr size_t OFF_H0   = 0;
static constexpr size_t OFF_M1   = OFF_H0 + alup((size_t)S0 * 4);
static constexpr size_t OFF_M2   = OFF_M1 + alup((size_t)S1 * 4);
static constexpr size_t OFF_M3   = OFF_M2 + alup((size_t)S2 * 4);
static constexpr size_t ABYTES   = alup((size_t)S1 * 16 * 4);   // 64 MiB arena
static constexpr size_t OFF_A1   = OFF_M3 + alup((size_t)S3 * 4);
static constexpr size_t OFF_A2   = OFF_A1 + ABYTES;
static constexpr size_t OFF_A3   = OFF_A2 + ABYTES;
static constexpr size_t OFF_A4   = OFF_A3 + ABYTES;
static constexpr size_t OFF_WP2B = OFF_A4 + ABYTES;
static constexpr size_t OFF_WP3A = OFF_WP2B + alup(27ul * 64 * 64 * 2);
static constexpr size_t OFF_WP3B = OFF_WP3A + alup(27ul * 64 * 512 * 2);
static constexpr size_t OFF_STAT = OFF_WP3B + alup(27ul * 512 * 512 * 2);
static constexpr size_t OFF_CNT  = OFF_STAT + alup(2368ul * 4);
static constexpr size_t OFF_POOL = OFF_CNT + alup(32ul * 4);

static constexpr int ilog2(int v) { return (v <= 1) ? 0 : 1 + ilog2(v >> 1); }

// ---------------- small utility kernels ----------------
__global__ void k_zero(float* __restrict__ p, long n) {
    long i = (long)blockIdx.x * blockDim.x + threadIdx.x;
    if (i < n) p[i] = 0.f;
}

// h0 = x * (mask != 0)
__global__ void k_h0(const float* __restrict__ x, const int* __restrict__ mask,
                     float* __restrict__ h0, long n) {
    long i = (long)blockIdx.x * blockDim.x + threadIdx.x;
    if (i < n) h0[i] = mask[i] ? x[i] : 0.f;
}

// 2x2x2 max-pool of an int mask -> float mask + active count (LDS reduce).
__global__ void k_pool_int(const int* __restrict__ mi, float* __restrict__ mo,
                           float* __restrict__ cnt, int Bn, int Din) {
    const int Dout = Din >> 1;
    const long spB = (long)Dout * Dout * Dout;
    const long total = spB * Bn;
    long idx = (long)blockIdx.x * blockDim.x + threadIdx.x;
    float v = 0.f;
    if (idx < total) {
        int b = (int)(idx / spB);
        long r = idx % spB;
        int z = (int)(r / (Dout * Dout)); int rr = (int)(r % (Dout * Dout));
        int y = rr / Dout, x = rr % Dout;
        for (int dz = 0; dz < 2; ++dz)
            for (int dy = 0; dy < 2; ++dy)
                for (int dx = 0; dx < 2; ++dx) {
                    long s = (((long)b * Din + 2 * z + dz) * Din + 2 * y + dy) * Din + 2 * x + dx;
                    if (mi[s]) v = 1.f;
                }
        mo[idx] = v;
    }
    __shared__ float sh[256];
    sh[threadIdx.x] = v; __syncthreads();
    for (int o = 128; o > 0; o >>= 1) {
        if (threadIdx.x < o) sh[threadIdx.x] += sh[threadIdx.x + o];
        __syncthreads();
    }
    if (threadIdx.x == 0 && sh[0] != 0.f) atomicAdd(cnt, sh[0]);
}

// 2x2x2 max-pool of a float mask -> float mask + total count + per-batch count.
__global__ void k_pool(const float* __restrict__ mi, float* __restrict__ mo,
                       float* __restrict__ cnt, float* __restrict__ cntB,
                       int Bn, int Din) {
    const int Dout = Din >> 1;
    const long spB = (long)Dout * Dout * Dout;
    const long total = spB * Bn;
    long idx = (long)blockIdx.x * blockDim.x + threadIdx.x;
    float v = 0.f; int b = 0;
    if (idx < total) {
        b = (int)(idx / spB);
        long r = idx % spB;
        int z = (int)(r / (Dout * Dout)); int rr = (int)(r % (Dout * Dout));
        int y = rr / Dout, x = rr % Dout;
        for (int dz = 0; dz < 2; ++dz)
            for (int dy = 0; dy < 2; ++dy)
                for (int dx = 0; dx < 2; ++dx) {
                    long s = (((long)b * Din + 2 * z + dz) * Din + 2 * y + dy) * Din + 2 * x + dx;
                    v = fmaxf(v, mi[s]);
                }
        mo[idx] = v;
    }
    __shared__ float sh[256];
    sh[threadIdx.x] = v; __syncthreads();
    for (int o = 128; o > 0; o >>= 1) {
        if (threadIdx.x < o) sh[threadIdx.x] += sh[threadIdx.x + o];
        __syncthreads();
    }
    if (threadIdx.x == 0 && sh[0] != 0.f) {
        atomicAdd(cnt, sh[0]);
        if (cntB) atomicAdd(&cntB[b], sh[0]);   // blocks never straddle batches (spB % 256 == 0)
    }
}

// Scalar masked conv3d, NDHWC x DHWIO, pad 1. One thread per (site, cout).
template <int STRIDE, int DIN, int DOUT, int CIN, int COUT>
__global__ void k_conv_scalar(const float* __restrict__ xin, const float* __restrict__ w,
                              const float* __restrict__ mout, float* __restrict__ yout) {
    constexpr int LD = ilog2(DOUT);
    constexpr int DM = DOUT - 1;
    long idx = (long)blockIdx.x * blockDim.x + threadIdx.x;
    constexpr long total = (long)BN_ * DOUT * DOUT * DOUT * COUT;
    if (idx >= total) return;
    int co = (int)(idx & (COUT - 1));
    long sidx = idx >> ilog2(COUT);
    if (mout[sidx] == 0.f) { yout[idx] = 0.f; return; }
    int b  = (int)(sidx >> (3 * LD));
    int z  = (int)(sidx >> (2 * LD)) & DM;
    int y  = (int)(sidx >> LD) & DM;
    int xx = (int)sidx & DM;
    float acc = 0.f;
    for (int t = 0; t < 27; ++t) {
        int dz = t / 9 - 1, dy = (t / 3) % 3 - 1, dx = t % 3 - 1;
        int iz = z * STRIDE + dz, iy = y * STRIDE + dy, ix = xx * STRIDE + dx;
        if ((unsigned)iz < (unsigned)DIN && (unsigned)iy < (unsigned)DIN &&
            (unsigned)ix < (unsigned)DIN) {
            long isite = (((long)b * DIN + iz) * DIN + iy) * DIN + ix;
            const float* xp = xin + isite * CIN;
            const float* wpt = w + (long)t * CIN * COUT + co;
#pragma unroll
            for (int ci = 0; ci < CIN; ++ci)
                acc = fmaf(xp[ci], wpt[(long)ci * COUT], acc);
        }
    }
    yout[idx] = acc;
}

// Pack f32 DHWIO weights into per-lane WMMA B-fragment layout:
// wp[(((t*KC + kc)*NT + nt)*32 + lane)*16 + j] = (f16) w[t][kc*32 + (lane>=16?8:0) + (j&7) + (j>=8?16:0)][nt*16 + (lane&15)]
__global__ void k_pack_w(const float* __restrict__ w, _Float16* __restrict__ wp,
                         int Cin, int Cout) {
    const int NT = Cout >> 4, KC = Cin >> 5;
    long total = 27L * KC * NT * 32 * 16;
    long idx = (long)blockIdx.x * blockDim.x + threadIdx.x;
    if (idx >= total) return;
    int j = (int)(idx & 15); long r = idx >> 4;
    int lane = (int)(r & 31); r >>= 5;
    int nt = (int)(r % NT); r /= NT;
    int kc = (int)(r % KC); int t = (int)(r / KC);
    int ci = kc * 32 + ((lane & 16) ? 8 : 0) + (j & 7) + ((j & 8) ? 16 : 0);
    int co = nt * 16 + (lane & 15);
    wp[idx] = (_Float16)w[((long)t * Cin + ci) * Cout + co];
}

// Implicit-GEMM WMMA conv: M = output sites, N = Cout, K = 27*Cin.
// Each wave owns NTPW adjacent 16x16 N-tiles of one M-tile; the A fragment
// is loaded once per K-chunk and reused for NTPW independent WMMAs.
// 4 waves per block cover 4*NTPW N-tiles.
template <int STRIDE, int DIN, int DOUT, int CIN, int COUT, int NTPW>
__global__ __launch_bounds__(128) void k_wmma_conv(
    const _Float16* __restrict__ xin, const _Float16* __restrict__ wp,
    const float* __restrict__ mout, float* __restrict__ yout) {
    constexpr int NT = COUT >> 4;
    constexpr int KC = CIN >> 5;
    constexpr int LD = ilog2(DOUT);
    constexpr int DM = DOUT - 1;
    const int lane = threadIdx.x & 31;
    const int wave = threadIdx.x >> 5;
    const int nt0 = (blockIdx.y * 4 + wave) * NTPW;
    const int mt = blockIdx.x;

    // Decode this lane's A-row (rows 0..15 duplicated across the two half-waves).
    const long s = (long)mt * 16 + (lane & 15);
    const int b  = (int)(s >> (3 * LD));
    const int z  = (int)(s >> (2 * LD)) & DM;
    const int y  = (int)(s >> LD) & DM;
    const int xx = (int)s & DM;
    const int base8 = (lane < 16) ? 0 : 8;   // ISA 16-bit A layout K-base
    const uint4 z4 = make_uint4(0u, 0u, 0u, 0u);

    // Running packed-weight pointer for this wave's first n-tile.
    const _Float16* wb = wp + (long)nt0 * 512 + lane * 16;
    constexpr long WTAP = (long)KC * NT * 512;   // halves per tap

    v8f acc[NTPW][2];
#pragma unroll
    for (int u = 0; u < NTPW; ++u) { acc[u][0] = {}; acc[u][1] = {}; }

    for (int t = 0; t < 27; ++t) {
        const int dz = t / 9 - 1, dy = (t / 3) % 3 - 1, dx = t % 3 - 1;
        const int iz = z * STRIDE + dz, iy = y * STRIDE + dy, ix = xx * STRIDE + dx;
        const bool inb = ((unsigned)iz < (unsigned)DIN) && ((unsigned)iy < (unsigned)DIN) &&
                         ((unsigned)ix < (unsigned)DIN);
        const _Float16* ab =
            xin + ((((long)b * DIN + iz) * DIN + iy) * DIN + ix) * CIN + base8;
        if (t < 26) {  // prefetch next tap's packed weights (global_prefetch_b8)
#pragma unroll
            for (int u = 0; u < NTPW; ++u)
                __builtin_prefetch(wb + WTAP + u * 512, 0, 1);
        }
#pragma unroll
        for (int kc = 0; kc < KC; ++kc) {
            F16Frag A;
            if (inb) {
                A.q[0] = *reinterpret_cast<const uint4*>(ab + kc * 32);
                A.q[1] = *reinterpret_cast<const uint4*>(ab + kc * 32 + 16);
            } else {
                A.q[0] = z4; A.q[1] = z4;
            }
#pragma unroll
            for (int u = 0; u < NTPW; ++u) {
                F16Frag Bf;
                const uint4* bp =
                    reinterpret_cast<const uint4*>(wb + (long)kc * NT * 512 + u * 512);
                Bf.q[0] = bp[0]; Bf.q[1] = bp[1];
                acc[u][kc & 1] = __builtin_amdgcn_wmma_f32_16x16x32_f16(
                    false, A.v, false, Bf.v, (short)0, acc[u][kc & 1], false, false);
            }
        }
        wb += WTAP;
    }
    const int mbase = (lane < 16) ? 0 : 8;   // ISA f32 C/D layout
#pragma unroll
    for (int u = 0; u < NTPW; ++u) {
        v8f a = acc[u][0] + acc[u][1];
        const int n = (nt0 + u) * 16 + (lane & 15);
#pragma unroll
        for (int v = 0; v < 8; ++v) {
            long sm = (long)mt * 16 + mbase + v;
            yout[sm * COUT + n] = a[v] * mout[sm];
        }
    }
}

// Per-channel sum / sumsq over all sites (inactive sites hold zeros).
__global__ void k_bn_stats(const float* __restrict__ x, float* __restrict__ sums,
                           long nsites, int C) {
    const int c = blockIdx.y;
    float s = 0.f, s2 = 0.f;
    for (long i = (long)blockIdx.x * blockDim.x + threadIdx.x; i < nsites;
         i += (long)gridDim.x * blockDim.x) {
        float v = x[i * C + c];
        s += v; s2 = fmaf(v, v, s2);
    }
    __shared__ float sh[512];
    sh[threadIdx.x] = s; sh[256 + threadIdx.x] = s2;
    __syncthreads();
    for (int o = 128; o > 0; o >>= 1) {
        if (threadIdx.x < o) {
            sh[threadIdx.x] += sh[threadIdx.x + o];
            sh[256 + threadIdx.x] += sh[256 + threadIdx.x + o];
        }
        __syncthreads();
    }
    if (threadIdx.x == 0) {
        atomicAdd(&sums[c], sh[0]);
        atomicAdd(&sums[C + c], sh[256]);
    }
}

// BN (active-site stats) + ELU + mask; optional f32 and f16 outputs.
__global__ void k_bn_apply(const float* __restrict__ xraw, const float* __restrict__ mout,
                           const float* __restrict__ sums, const float* __restrict__ cnt,
                           const float* __restrict__ gamma, const float* __restrict__ beta,
                           float* __restrict__ yf32, _Float16* __restrict__ yf16,
                           long nsites, int C) {
    long idx = (long)blockIdx.x * blockDim.x + threadIdx.x;
    long total = nsites * C;
    if (idx >= total) return;
    int c = (int)(idx % C);
    long sidx = idx / C;
    float n = cnt[0];
    float mean = sums[c] / n;
    float var = sums[C + c] / n - mean * mean;
    float xh = (xraw[idx] - mean) * rsqrtf(var + 1e-5f) * gamma[c] + beta[c];
    float e = xh > 0.f ? xh : expm1f(xh);
    e *= mout[sidx];
    if (yf32) yf32[idx] = e;
    if (yf16) yf16[idx] = (_Float16)e;
}

// Masked global mean pool: pooled[b,c] = sum_sites h[b,s,c] / cntB[b].
__global__ void k_pool_global(const float* __restrict__ h, const float* __restrict__ cntB,
                              float* __restrict__ pooled, int spB, int C, int Bn) {
    int idx = blockIdx.x * blockDim.x + threadIdx.x;
    if (idx >= Bn * C) return;
    int b = idx / C, c = idx % C;
    const float* p = h + (long)b * spB * C + c;
    float s = 0.f;
    for (int i = 0; i < spB; ++i) s += p[(long)i * C];
    pooled[idx] = s / cntB[b];
}

// mean = pooled @ wm + bm ; log_var = pooled @ wv + bv (concatenated in d_out).
__global__ void k_heads(const float* __restrict__ pooled, const float* __restrict__ wm,
                        const float* __restrict__ bm, const float* __restrict__ wv,
                        const float* __restrict__ bv, float* __restrict__ out,
                        int Bn, int C) {
    int idx = blockIdx.x * blockDim.x + threadIdx.x;
    if (idx >= Bn * C) return;
    int b = idx / C, c = idx % C;
    const float* p = pooled + (long)b * C;
    float sm = 0.f, sv = 0.f;
    for (int k = 0; k < C; ++k) {
        float pk = p[k];
        sm = fmaf(pk, wm[(long)k * C + c], sm);
        sv = fmaf(pk, wv[(long)k * C + c], sv);
    }
    out[idx] = sm + bm[c];
    out[(long)Bn * C + idx] = sv + bv[c];
}

// ---------------------------------------------------------------------------
extern "C" void kernel_launch(void* const* d_in, const int* in_sizes, int n_in,
                              void* d_out, int out_size, void* d_ws, size_t ws_size,
                              hipStream_t stream) {
    (void)in_sizes; (void)n_in; (void)out_size; (void)ws_size;
    const float* x    = (const float*)d_in[0];
    const int*   mask = (const int*)d_in[1];
    const float* w1a = (const float*)d_in[2],  *g1a = (const float*)d_in[3],  *b1a = (const float*)d_in[4];
    const float* w1b = (const float*)d_in[5],  *g1b = (const float*)d_in[6],  *b1b = (const float*)d_in[7];
    const float* w2a = (const float*)d_in[8],  *g2a = (const float*)d_in[9],  *b2a = (const float*)d_in[10];
    const float* w2b = (const float*)d_in[11], *g2b = (const float*)d_in[12], *b2b = (const float*)d_in[13];
    const float* w3a = (const float*)d_in[14], *g3a = (const float*)d_in[15], *b3a = (const float*)d_in[16];
    const float* w3b = (const float*)d_in[17], *g3b = (const float*)d_in[18], *b3b = (const float*)d_in[19];
    const float* wm  = (const float*)d_in[20], *bm  = (const float*)d_in[21];
    const float* wv  = (const float*)d_in[22], *bv  = (const float*)d_in[23];
    float* out = (float*)d_out;

    char* ws = (char*)d_ws;
    float*     h0   = (float*)(ws + OFF_H0);
    float*     m1   = (float*)(ws + OFF_M1);
    float*     m2   = (float*)(ws + OFF_M2);
    float*     m3   = (float*)(ws + OFF_M3);
    float*     A1   = (float*)(ws + OFF_A1);
    float*     A2   = (float*)(ws + OFF_A2);
    float*     A3   = (float*)(ws + OFF_A3);
    float*     A4   = (float*)(ws + OFF_A4);
    _Float16*  wp2b = (_Float16*)(ws + OFF_WP2B);
    _Float16*  wp3a = (_Float16*)(ws + OFF_WP3A);
    _Float16*  wp3b = (_Float16*)(ws + OFF_WP3B);
    float*     stats = (float*)(ws + OFF_STAT);
    float* st1a = stats + 0,   *st1b = stats + 32;
    float* st2a = stats + 64,  *st2b = stats + 192;
    float* st3a = stats + 320, *st3b = stats + 1344;
    float* cnt  = (float*)(ws + OFF_CNT);      // [0]=cnt1 [1]=cnt2 [2]=cnt3 [4..7]=cntB
    float* cntB = cnt + 4;
    float* pooled = (float*)(ws + OFF_POOL);

    const dim3 T(256);
    auto G = [](long n) { return dim3((unsigned)((n + 255) / 256)); };

    // --- reset reduction buffers (graph-replay safe) ---
    k_zero<<<G(2368), T, 0, stream>>>(stats, 2368);
    k_zero<<<G(32),   T, 0, stream>>>(cnt, 32);

    // --- input masking + mask pyramid level 1 ---
    k_h0<<<G(S0), T, 0, stream>>>(x, mask, h0, S0);
    k_pool_int<<<G(S1), T, 0, stream>>>(mask, m1, &cnt[0], BN_, D0);

    // --- block 1 (scalar path: Cin 1 / 16) ---
    k_conv_scalar<2, 128, 64, 1, 16><<<G(S1 * 16), T, 0, stream>>>(h0, w1a, m1, A1);
    k_bn_stats<<<dim3(64, 16), T, 0, stream>>>(A1, st1a, S1, 16);
    k_bn_apply<<<G(S1 * 16), T, 0, stream>>>(A1, m1, st1a, &cnt[0], g1a, b1a, A2, nullptr, S1, 16);
    k_conv_scalar<1, 64, 64, 16, 16><<<G(S1 * 16), T, 0, stream>>>(A2, w1b, m1, A3);
    k_bn_stats<<<dim3(64, 16), T, 0, stream>>>(A3, st1b, S1, 16);
    k_bn_apply<<<G(S1 * 16), T, 0, stream>>>(A3, m1, st1b, &cnt[0], g1b, b1b, A4, nullptr, S1, 16);

    // --- block 2 (2a scalar, 2b WMMA) ---
    k_pool<<<G(S2), T, 0, stream>>>(m1, m2, &cnt[1], nullptr, BN_, D1);
    k_conv_scalar<2, 64, 32, 16, 64><<<G(S2 * 64), T, 0, stream>>>(A4, w2a, m2, A1);
    k_bn_stats<<<dim3(64, 64), T, 0, stream>>>(A1, st2a, S2, 64);
    k_bn_apply<<<G(S2 * 64), T, 0, stream>>>(A1, m2, st2a, &cnt[1], g2a, b2a, nullptr,
                                             (_Float16*)A2, S2, 64);
    k_pack_w<<<G(27L * 64 * 64), T, 0, stream>>>(w2b, wp2b, 64, 64);
    k_wmma_conv<1, 32, 32, 64, 64, 1>
        <<<dim3((unsigned)(S2 / 16), 1), dim3(128), 0, stream>>>(
            (const _Float16*)A2, wp2b, m2, A3);
    k_bn_stats<<<dim3(64, 64), T, 0, stream>>>(A3, st2b, S2, 64);
    k_bn_apply<<<G(S2 * 64), T, 0, stream>>>(A3, m2, st2b, &cnt[1], g2b, b2b, nullptr,
                                             (_Float16*)A4, S2, 64);

    // --- block 3 (both convs WMMA, 2 N-tiles per wave) ---
    k_pool<<<G(S3), T, 0, stream>>>(m2, m3, &cnt[2], cntB, BN_, D2);
    k_pack_w<<<G(27L * 64 * 512), T, 0, stream>>>(w3a, wp3a, 64, 512);
    k_wmma_conv<2, 32, 16, 64, 512, 2>
        <<<dim3((unsigned)(S3 / 16), 4), dim3(128), 0, stream>>>(
            (const _Float16*)A4, wp3a, m3, A1);
    k_bn_stats<<<dim3(64, 512), T, 0, stream>>>(A1, st3a, S3, 512);
    k_bn_apply<<<G(S3 * 512), T, 0, stream>>>(A1, m3, st3a, &cnt[2], g3a, b3a, nullptr,
                                              (_Float16*)A2, S3, 512);
    k_pack_w<<<G(27L * 512 * 512), T, 0, stream>>>(w3b, wp3b, 512, 512);
    k_wmma_conv<1, 16, 16, 512, 512, 2>
        <<<dim3((unsigned)(S3 / 16), 4), dim3(128), 0, stream>>>(
            (const _Float16*)A2, wp3b, m3, A3);
    k_bn_stats<<<dim3(64, 512), T, 0, stream>>>(A3, st3b, S3, 512);
    k_bn_apply<<<G(S3 * 512), T, 0, stream>>>(A3, m3, st3b, &cnt[2], g3b, b3b, A4, nullptr,
                                              S3, 512);

    // --- global masked mean pool + linear heads ---
    k_pool_global<<<G(BN_ * 512), T, 0, stream>>>(A4, cntB, pooled, D3 * D3 * D3, 512, BN_);
    k_heads<<<G(BN_ * 512), T, 0, stream>>>(pooled, wm, bm, wv, bv, out, BN_, 512);
}
